// AttentionMixerRec_42734924595928
// MI455X (gfx1250) — matrix operand order repaired and checked by reading, not compiled, not run on hardware
//
#include <hip/hip_runtime.h>
#include <hip/hip_bf16.h>

// Problem constants (match reference)
#define B_ 256
#define S_ 200
#define D_ 256
#define L_ 5
#define H_ 4
#define HL_ 20      // H*L rows of folded query matrix
#define RROWS_ 32   // HL_ padded to WMMA M-tile multiple
#define DD_ (D_*D_)

// CDNA5 async global->LDS copy path (guarded; falls back to sync staging)
#if defined(__has_builtin)
#  if __has_builtin(__builtin_amdgcn_global_load_async_to_lds_b128) && \
      __has_builtin(__builtin_amdgcn_s_wait_asynccnt)
#    define USE_ASYNC_LDS 1
#  endif
#endif
#ifndef USE_ASYNC_LDS
#  define USE_ASYNC_LDS 0
#endif

typedef __attribute__((ext_vector_type(16))) __bf16 v16bf;
typedef __attribute__((ext_vector_type(8)))  float  v8f;

// 128-bit payload type for async LDS copies (matches builtin signature)
typedef int v4i_vs __attribute__((vector_size(16)));
typedef __attribute__((address_space(1))) v4i_vs glob_b128_t;
typedef __attribute__((address_space(3))) v4i_vs lds_b128_t;

// ---------------- bf16 helpers (round-to-nearest-even) ----------------
__device__ __forceinline__ unsigned short f2bf(float f) {
  unsigned u = __float_as_uint(f);
  u += 0x7FFFu + ((u >> 16) & 1u);
  return (unsigned short)(u >> 16);
}
__device__ __forceinline__ unsigned short to_bf16_bits(float f)         { return f2bf(f); }
__device__ __forceinline__ unsigned short to_bf16_bits(unsigned short b){ return b; }
__device__ __forceinline__ void store_out_elem(float* p, float v)          { *p = v; }
__device__ __forceinline__ void store_out_elem(unsigned short* p, float v) { *p = f2bf(v); }

// ---------------- 1) gather: emb_bf16[b,s,:] = bf16(table[seq[b,s]]) ----------------
__global__ __launch_bounds__(D_) void gather_kernel(
    const int* __restrict__ seq, const float* __restrict__ table,
    unsigned short* __restrict__ embbf) {
  long bs = blockIdx.x;                 // b*S_ + s
  int d = threadIdx.x;
  long idx = seq[bs];
  embbf[bs * D_ + d] = f2bf(table[idx * D_ + d]);
}

// ---------------- 2) q0[b,l,:] = sum of emb rows at unique last-(l+1) positions ------
__global__ __launch_bounds__(D_) void q0_kernel(
    const int* __restrict__ seq, const int* __restrict__ seqlen,
    const float* __restrict__ table, float* __restrict__ q0) {
  int b = blockIdx.x;
  int d = threadIdx.x;
  int len = seqlen[b];
  float acc = 0.f;
  int prev = -1;
  #pragma unroll
  for (int l = 0; l < L_; ++l) {
    int p = len - 1 - l; if (p < 0) p = 0;
    if (p != prev) {                    // positions strictly decrease until clamped at 0
      long idx = seq[(long)b * S_ + p];
      acc += table[idx * D_ + d];
    }
    prev = p;
    q0[((long)b * L_ + l) * D_ + d] = acc;
  }
}

// ============ generic tiled bf16 WMMA GEMM =================================
// Block: 256 threads = 8 waves; block tile 32(M) x 128(N); K step 32.
// Wave w: rows wm=(w>>2)*16, cols wn=(w&3)*32 -> 2 WMMAs/K-step sharing A frag.
// Per-batch (blockIdx.z) operand offsets: off = (z % mod)*s1 + (z / mod)*s2.
template<typename TA, typename TB, typename TO>
__global__ __launch_bounds__(256) void gemm_bf16_wmma(
    const TA* __restrict__ Abase, const TB* __restrict__ Bbase, TO* __restrict__ Obase,
    int N, int K, int lda, int ldb, int ldo,
    int transA, int transB,
    int modA, long long sA1, long long sA2,
    int modB, long long sB1, long long sB2,
    int modO, long long sO1, long long sO2,
    int Mstore, int Nstore, float scale) {
  int z = blockIdx.z;
  const TA* A = Abase + (long long)(z % modA) * sA1 + (long long)(z / modA) * sA2;
  const TB* B = Bbase + (long long)(z % modB) * sB1 + (long long)(z / modB) * sB2;
  TO*       O = Obase + (long long)(z % modO) * sO1 + (long long)(z / modO) * sO2;

  int mBlk = blockIdx.y * 32;
  int nBlk = blockIdx.x * 128;
  int tid  = threadIdx.x;
  unsigned lane = tid & 31u;
  unsigned wave = tid >> 5;              // 0..7
  unsigned wm = (wave >> 2) * 16u;       // 0 / 16
  unsigned wn = (wave & 3u) * 32u;       // 0/32/64/96
  unsigned hh = lane >> 4;               // lane half (0 or 1)
  unsigned mr = lane & 15u;

  // +2 shorts pad -> 68B row stride (17 banks, odd) for conflict-free frag reads
  __shared__ unsigned short Asm[32][34];    // [m][k]
  __shared__ unsigned short Bsm[128][34];   // [n][k] (k-contiguous per column)

  v8f acc0 = {0.f,0.f,0.f,0.f,0.f,0.f,0.f,0.f};
  v8f acc1 = {0.f,0.f,0.f,0.f,0.f,0.f,0.f,0.f};

  for (int k0 = 0; k0 < K; k0 += 32) {
    // A tile: 32x32 = 1024 elems, 4/thread (coalesced in k or m)
    #pragma unroll
    for (int i = 0; i < 4; ++i) {
      int li = i * 256 + tid;
      int r = li >> 5, c = li & 31;
      long long gm = mBlk + r, gk = k0 + c;
      Asm[r][c] = to_bf16_bits(transA ? A[gk * lda + gm] : A[gm * lda + gk]);
    }
    // B tile: 32x128 = 4096 elems, 16/thread (coalesced over n when !transB)
    #pragma unroll
    for (int i = 0; i < 16; ++i) {
      int li = i * 256 + tid;
      int kk = li >> 7, n = li & 127;
      long long gk = k0 + kk;
      long long gn = nBlk + n; if (gn >= N) gn = N - 1;   // clamp ragged N
      Bsm[n][kk] = to_bf16_bits(transB ? B[gn * ldb + gk] : B[gk * ldb + gn]);
    }
    __syncthreads();

    // WMMA fragments in contiguous k order (k = 16*half + s), same for A and B.
    union { v16bf v; unsigned short u[16]; } af, bf0, bf1;
    #pragma unroll
    for (int j = 0; j < 8; ++j) {
      // A 16x32: lanes0-15 K0..15; lanes16-31 K16..31
      int ka = 2 * j + 16 * (int)hh;
      af.u[2*j]   = Asm[wm + mr][ka];
      af.u[2*j+1] = Asm[wm + mr][ka + 1];
      // B 32x16: lane n (0-15) col n K0..15; lane n+16 col n K16..31
      int kb = 2 * j + 16 * (int)hh;
      bf0.u[2*j]   = Bsm[wn + mr][kb];
      bf0.u[2*j+1] = Bsm[wn + mr][kb + 1];
      bf1.u[2*j]   = Bsm[wn + 16u + mr][kb];
      bf1.u[2*j+1] = Bsm[wn + 16u + mr][kb + 1];
    }
    acc0 = __builtin_amdgcn_wmma_f32_16x16x32_bf16(false, af.v, false, bf0.v,
                                                   (short)0, acc0, false, false);
    acc1 = __builtin_amdgcn_wmma_f32_16x16x32_bf16(false, af.v, false, bf1.v,
                                                   (short)0, acc1, false, false);
    __syncthreads();
  }

  // C/D 16x16 f32 layout: VGPR v, lanes0-15 -> (M=v,N=lane); lanes16-31 -> (M=v+8)
  #pragma unroll
  for (int v = 0; v < 8; ++v) {
    int m = mBlk + (int)wm + (int)(hh * 8u) + v;
    int n0 = nBlk + (int)wn + (int)mr;
    if (m < Mstore) {
      if (n0 < Nstore)      store_out_elem(&O[(long long)m * ldo + n0],      acc0[v] * scale);
      if (n0 + 16 < Nstore) store_out_elem(&O[(long long)m * ldo + n0 + 16], acc1[v] * scale);
    }
  }
}

// ============ specialized per-batch scores GEMM: async DMA + double buffer ==
// scores[b] (20x200) = r[b] (32xD) @ emb[b]^T (DxS).  Both operands bf16 and
// 16B-contiguous on both sides -> GLOBAL_LOAD_ASYNC_TO_LDS_B128, with LDS
// double buffering so tile k+1's DMA overlaps tile k's WMMAs.
__global__ __launch_bounds__(256) void scores_gemm_kernel(
    const unsigned short* __restrict__ rbf, const unsigned short* __restrict__ embbf,
    float* __restrict__ scor) {
  int b = blockIdx.z;
  const unsigned short* A  = rbf   + (long long)b * RROWS_ * D_;
  const unsigned short* Bm = embbf + (long long)b * S_ * D_;
  float*                O  = scor  + (long long)b * HL_ * S_;

  int nBlk = blockIdx.x * 128;
  int tid  = threadIdx.x;
  unsigned lane = tid & 31u;
  unsigned wave = tid >> 5;
  unsigned wm = (wave >> 2) * 16u;
  unsigned wn = (wave & 3u) * 32u;
  unsigned hh = lane >> 4;
  unsigned mr = lane & 15u;

  // 40-short (80B) row stride keeps every 8-short chunk 16B aligned for B128.
  __shared__ unsigned short Asm[2][32][40];    // [buf][m][k]
  __shared__ unsigned short Bsm[2][128][40];   // [buf][s][k]

  v8f acc0 = {0.f,0.f,0.f,0.f,0.f,0.f,0.f,0.f};
  v8f acc1 = {0.f,0.f,0.f,0.f,0.f,0.f,0.f,0.f};

#if USE_ASYNC_LDS
  // Per-thread copy assignments (fixed across iterations):
  //  A tile: 32 rows x 4 chunks of 8 shorts -> threads 0..127, one B128 each
  //  B tile: 128 rows x 4 chunks           -> 512 copies, 2 per thread
  int ar = tid >> 2, ac = tid & 3;
  int bn0 = tid >> 2,           bc0 = tid & 3;
  int bn1 = (256 + tid) >> 2,   bc1 = tid & 3;
  int gn0 = nBlk + bn0; if (gn0 >= S_) gn0 = S_ - 1;
  int gn1 = nBlk + bn1; if (gn1 >= S_) gn1 = S_ - 1;

  #define ISSUE_TILE(kk, buf)                                                  \
    do {                                                                       \
      if (tid < 128)                                                           \
        __builtin_amdgcn_global_load_async_to_lds_b128(                        \
            (glob_b128_t*)(A + (long long)ar * D_ + (kk) + ac * 8),            \
            (lds_b128_t*)(&Asm[(buf)][ar][ac * 8]), 0, 0);                     \
      __builtin_amdgcn_global_load_async_to_lds_b128(                          \
          (glob_b128_t*)(Bm + (long long)gn0 * D_ + (kk) + bc0 * 8),           \
          (lds_b128_t*)(&Bsm[(buf)][bn0][bc0 * 8]), 0, 0);                     \
      __builtin_amdgcn_global_load_async_to_lds_b128(                          \
          (glob_b128_t*)(Bm + (long long)gn1 * D_ + (kk) + bc1 * 8),           \
          (lds_b128_t*)(&Bsm[(buf)][bn1][bc1 * 8]), 0, 0);                     \
    } while (0)

  // Prologue: fill buffer 0 with tile k=0.
  ISSUE_TILE(0, 0);
  __builtin_amdgcn_s_wait_asynccnt(0);
  __syncthreads();

  int cur = 0;
  for (int k0 = 0; k0 < D_; k0 += 32) {
    // Prefetch next K tile into the other buffer (overlaps with WMMAs below).
    if (k0 + 32 < D_) ISSUE_TILE(k0 + 32, cur ^ 1);

    union { v16bf v; unsigned short u[16]; } af, bf0, bf1;
    #pragma unroll
    for (int j = 0; j < 8; ++j) {
      int ka = 2 * j + 16 * (int)hh;
      af.u[2*j]   = Asm[cur][wm + mr][ka];
      af.u[2*j+1] = Asm[cur][wm + mr][ka + 1];
      int kb = 2 * j + 16 * (int)hh;
      bf0.u[2*j]   = Bsm[cur][wn + mr][kb];
      bf0.u[2*j+1] = Bsm[cur][wn + mr][kb + 1];
      bf1.u[2*j]   = Bsm[cur][wn + 16u + mr][kb];
      bf1.u[2*j+1] = Bsm[cur][wn + 16u + mr][kb + 1];
    }
    acc0 = __builtin_amdgcn_wmma_f32_16x16x32_bf16(false, af.v, false, bf0.v,
                                                   (short)0, acc0, false, false);
    acc1 = __builtin_amdgcn_wmma_f32_16x16x32_bf16(false, af.v, false, bf1.v,
                                                   (short)0, acc1, false, false);

    // Next buffer complete (own wave) + all waves done reading/writing.
    __builtin_amdgcn_s_wait_asynccnt(0);
    __syncthreads();
    cur ^= 1;
  }
  #undef ISSUE_TILE
#else
  for (int k0 = 0; k0 < D_; k0 += 32) {
    #pragma unroll
    for (int i = 0; i < 4; ++i) {
      int li = i * 256 + tid;
      int r = li >> 5, c = li & 31;
      Asm[0][r][c] = A[(long long)r * D_ + k0 + c];
    }
    #pragma unroll
    for (int i = 0; i < 16; ++i) {
      int li = i * 256 + tid;
      int kk = li >> 7, n = li & 127;
      int gn = nBlk + n; if (gn >= S_) gn = S_ - 1;
      Bsm[0][n][kk] = Bm[(long long)gn * D_ + k0 + kk];
    }
    __syncthreads();
    union { v16bf v; unsigned short u[16]; } af, bf0, bf1;
    #pragma unroll
    for (int j = 0; j < 8; ++j) {
      int ka = 2 * j + 16 * (int)hh;
      af.u[2*j]   = Asm[0][wm + mr][ka];
      af.u[2*j+1] = Asm[0][wm + mr][ka + 1];
      int kb = 2 * j + 16 * (int)hh;
      bf0.u[2*j]   = Bsm[0][wn + mr][kb];
      bf0.u[2*j+1] = Bsm[0][wn + mr][kb + 1];
      bf1.u[2*j]   = Bsm[0][wn + 16u + mr][kb];
      bf1.u[2*j+1] = Bsm[0][wn + 16u + mr][kb + 1];
    }
    acc0 = __builtin_amdgcn_wmma_f32_16x16x32_bf16(false, af.v, false, bf0.v,
                                                   (short)0, acc0, false, false);
    acc1 = __builtin_amdgcn_wmma_f32_16x16x32_bf16(false, af.v, false, bf1.v,
                                                   (short)0, acc1, false, false);
    __syncthreads();
  }
#endif

  #pragma unroll
  for (int v = 0; v < 8; ++v) {
    int m = (int)wm + (int)(hh * 8u) + v;
    int n0 = nBlk + (int)wn + (int)mr;
    if (m < HL_) {
      if (n0 < S_)      O[(long long)m * S_ + n0]      = acc0[v];
      if (n0 + 16 < S_) O[(long long)m * S_ + n0 + 16] = acc1[v];
    }
  }
}

// ---------------- zero the padding rows (hl = 20..31) of r ----------------
__global__ __launch_bounds__(256) void zero_rpad_kernel(unsigned short* __restrict__ rbf) {
  long i = (long)blockIdx.x * 256 + threadIdx.x;    // over B_*12*D_
  long b = i / (12 * D_);
  long rem = i % (12 * D_);
  long row = HL_ + rem / D_;
  long d = rem % D_;
  rbf[(b * RROWS_ + row) * D_ + d] = 0;             // bf16 +0.0
}

// ---------------- softmax over S + p=4 pool over L + head mean ----------------
__global__ __launch_bounds__(256) void softmax_pool_kernel(
    const float* __restrict__ scores, float* __restrict__ pooled) {
  int b = blockIdx.x, t = threadIdx.x;
  __shared__ float red[256];
  bool act = (t < S_);
  float pool = 0.f;
  for (int h = 0; h < H_; ++h) {
    float acc4 = 0.f;
    for (int l = 0; l < L_; ++l) {
      int row = h * L_ + l;
      float sc = act ? scores[((long)b * HL_ + row) * S_ + t] : -1e30f;
      red[t] = sc; __syncthreads();
      for (int o = 128; o > 0; o >>= 1) { if (t < o) red[t] = fmaxf(red[t], red[t + o]); __syncthreads(); }
      float mx = red[0]; __syncthreads();
      float e = act ? __expf(sc - mx) : 0.f;
      red[t] = e; __syncthreads();
      for (int o = 128; o > 0; o >>= 1) { if (t < o) red[t] += red[t + o]; __syncthreads(); }
      float sum = red[0]; __syncthreads();
      float a = e / sum;
      float a2 = a * a;
      acc4 += a2 * a2;
    }
    pool += sqrtf(sqrtf(acc4));
  }
  if (act) pooled[(long)b * S_ + t] = pool * (1.0f / (float)H_);
}

// ---------------- out[b,d] = sum_s pooled[b,s] * table[seq[b,s], d] (fp32) ----------
__global__ __launch_bounds__(D_) void out_kernel(
    const int* __restrict__ seq, const float* __restrict__ table,
    const float* __restrict__ pooled, float* __restrict__ out) {
  int b = blockIdx.x, d = threadIdx.x;
  __shared__ float pw[S_];
  __shared__ int idxs[S_];
  for (int s = d; s < S_; s += D_) {
    pw[s] = pooled[(long)b * S_ + s];
    idxs[s] = seq[(long)b * S_ + s];
  }
  __syncthreads();
  float acc = 0.f;
  for (int s = 0; s < S_; ++s)
    acc = fmaf(pw[s], table[(long)idxs[s] * D_ + d], acc);
  out[(long)b * D_ + d] = acc;
}

// ---------------- launcher ----------------
extern "C" void kernel_launch(void* const* d_in, const int* in_sizes, int n_in,
                              void* d_out, int out_size, void* d_ws, size_t ws_size,
                              hipStream_t stream) {
  const int*   item_seq     = (const int*)d_in[0];
  const int*   item_seq_len = (const int*)d_in[1];
  const float* emb_table    = (const float*)d_in[2];
  const float* lin_W        = (const float*)d_in[3];
  const float* WQ_W         = (const float*)d_in[4];
  const float* WK_W         = (const float*)d_in[5];
  float* out = (float*)d_out;

  char* ws = (char*)d_ws;
  unsigned short* embBf = (unsigned short*)ws;  ws += (size_t)B_ * S_ * D_ * 2;      // 26.2 MB
  float*          q0    = (float*)ws;           ws += (size_t)B_ * L_ * D_ * 4;      // 1.3 MB
  float*          Cf    = (float*)ws;           ws += (size_t)H_ * DD_ * 4;          // 1.0 MB
  unsigned short* A2bf  = (unsigned short*)ws;  ws += (size_t)HL_ * DD_ * 2;         // 2.6 MB
  unsigned short* rbf   = (unsigned short*)ws;  ws += (size_t)B_ * RROWS_ * D_ * 2;  // 4.2 MB
  float*          scor  = (float*)ws;           ws += (size_t)B_ * HL_ * S_ * 4;     // 4.1 MB
  float*          pool  = (float*)ws;           ws += (size_t)B_ * S_ * 4;           // 0.2 MB

  // Stage 1: gather embeddings to bf16, and pooled queries q0 (fp32, exact).
  gather_kernel<<<B_ * S_, D_, 0, stream>>>(item_seq, emb_table, embBf);
  q0_kernel<<<B_, D_, 0, stream>>>(item_seq, item_seq_len, emb_table, q0);

  // GEMM1: C[h] = WQ[h]^T @ WK[h]   (4 x 256^3, batch-independent fold)
  gemm_bf16_wmma<float, float, float><<<dim3(2, 8, H_), 256, 0, stream>>>(
      WQ_W, WK_W, Cf, /*N=*/D_, /*K=*/D_, D_, D_, D_,
      /*transA=*/1, /*transB=*/0,
      1, 0, DD_,   1, 0, DD_,   1, 0, DD_,
      D_, D_, 1.0f);

  // GEMM2: A2[h,l] = lin[l]^T @ C[h]  (20 x 256^3), store bf16
  gemm_bf16_wmma<float, float, unsigned short><<<dim3(2, 8, HL_), 256, 0, stream>>>(
      lin_W, Cf, A2bf, D_, D_, D_, D_, D_,
      /*transA=*/1, /*transB=*/0,
      L_, DD_, 0,   L_, 0, DD_,   1, 0, DD_,
      D_, D_, 1.0f);

  // GEMM3: r[:,hl,:] = q0[:,l,:] @ A2[h,l] * (1/sqrt(D))  (20 x B*256^2), bf16 out
  zero_rpad_kernel<<<(B_ * 12 * D_) / 256, 256, 0, stream>>>(rbf);
  gemm_bf16_wmma<float, unsigned short, unsigned short><<<dim3(2, 8, HL_), 256, 0, stream>>>(
      q0, A2bf, rbf, D_, D_, /*lda=*/L_ * D_, D_, /*ldo=*/RROWS_ * D_,
      /*transA=*/0, /*transB=*/0,
      L_, D_, 0,   1, 0, DD_,   1, 0, D_,
      B_, D_, 1.0f / 16.0f);

  // GEMM4 (hot, per-b): scores[b] = r[b] @ emb[b]^T, async DMA double-buffered
  scores_gemm_kernel<<<dim3(2, 1, B_), 256, 0, stream>>>(rbf, embBf, scor);

  // Softmax over S, p=4 pool over L, mean over heads.
  softmax_pool_kernel<<<B_, 256, 0, stream>>>(scor, pool);

  // Final fp32 weighted embedding sum.
  out_kernel<<<B_, D_, 0, stream>>>(item_seq, emb_table, pool, out);
}
